// BahdanauAttention_1314259993035
// MI455X (gfx1250) — matrix operand order repaired and checked by reading, not compiled, not run on hardware
//
#include <hip/hip_runtime.h>
#include <hip/hip_bf16.h>
#include <math.h>

// Problem sizes (fixed by the reference)
#define B_  64
#define T_  2048
#define U_  512
#define QS_ 512
#define MS_ 512

#define KSTEP  32          // K per WMMA step
#define MTILE  64          // t-rows per block
#define ROWPAD 36          // floats per LDS row after TDM padding (32 + 4)

typedef __attribute__((ext_vector_type(16))) __bf16 v16bf;
typedef __attribute__((ext_vector_type(8)))  float  v8f;
typedef __attribute__((ext_vector_type(4)))  unsigned int u32x4;
typedef __attribute__((ext_vector_type(8)))  int          i32x8;
typedef __attribute__((ext_vector_type(4)))  int          i32x4;

union BF16Frag {
    v16bf  v;
    __bf16 e[16];
    int4   q[2];
};

// ---------------------------------------------------------------------------
// TDM: DMA a [64 rows x 32 cols] fp32 tile of keys (row stride 512 floats)
// into LDS at byte offset lds_off, padding 4 DWORDs per 32-DWORD row so the
// LDS row stride is 36 floats (conflict-free for stride-row reads).
// D# built per CDNA5 ISA ch.8 (groups 0..3); 2-D tensor -> groups 2/3 zero.
// This toolchain exposes the 6-arg builtin (extra int32x8 group + cpol).
// ---------------------------------------------------------------------------
__device__ __forceinline__ void tdm_load_keys_tile(const float* gsrc, unsigned lds_off)
{
    unsigned long long ga = (unsigned long long)(uintptr_t)gsrc;
    u32x4 g0;
    g0[0] = 1u;                                         // count=1 (valid), user D#
    g0[1] = lds_off;                                    // lds_addr (bytes)
    g0[2] = (unsigned)(ga & 0xFFFFFFFFu);               // global_addr[31:0]
    g0[3] = (unsigned)((ga >> 32) & 0x01FFFFFFu)        // global_addr[56:32]
          | (2u << 30);                                 // type = 2 ("image")
    i32x8 g1;
    g1[0] = (2 << 16)                                   // data_size = 4 bytes
          | (1 << 20)                                   // pad_enable
          | (4 << 22)                                   // pad_interval: 32 DWORDs
          | (3 << 25);                                  // pad_amount:   4 DWORDs
    g1[1] = (int)(512u << 16);                          // tensor_dim0 = 512 (lo 16 in [63:48])
    g1[2] = 0;                                          // tensor_dim0 hi / tensor_dim1 lo
    g1[3] = 2 | (KSTEP << 16);                          // tensor_dim1 = 131072 (hi), tile_dim0 = 32
    g1[4] = MTILE;                                      // tile_dim1 = 64 rows, tile_dim2 = 0
    g1[5] = 512;                                        // tensor_dim0_stride = 512 elements
    g1[6] = 0;
    g1[7] = 0;
    i32x4 z4 = { 0, 0, 0, 0 };                          // groups 2/3 unused (2-D)
    i32x8 z8 = { 0, 0, 0, 0, 0, 0, 0, 0 };              // extra group (unused)
    __builtin_amdgcn_tensor_load_to_lds(g0, g1, z4, z4, z8, 0);
}

// ---------------------------------------------------------------------------
// Kernel 0: convert Wm (fp32 [U, MS]) -> bf16 copy in workspace (L2-resident).
// ---------------------------------------------------------------------------
__global__ __launch_bounds__(256)
void convert_wm_kernel(const float* __restrict__ Wm, __bf16* __restrict__ out)
{
    const int i = blockIdx.x * 256 + threadIdx.x;
    out[i] = (__bf16)Wm[i];
}

// ---------------------------------------------------------------------------
// Kernel 1: pq[b,u] = sum_q query[b,q] * Wq[u,q].
// ---------------------------------------------------------------------------
__global__ __launch_bounds__(256)
void pq_kernel(const float* __restrict__ query,
               const float* __restrict__ Wq,
               float* __restrict__ pq)
{
    const int idx = blockIdx.x * 256 + threadIdx.x;   // 32768 = B_*U_
    const int b = idx >> 9;
    const int u = idx & (U_ - 1);
    const float4* q  = (const float4*)(query + (size_t)b * QS_);
    const float4* wv = (const float4*)(Wq    + (size_t)u * QS_);
    float acc = 0.0f;
#pragma unroll 8
    for (int i = 0; i < QS_ / 4; ++i) {
        float4 a = q[i], c = wv[i];
        acc += a.x * c.x + a.y * c.y + a.z * c.z + a.w * c.w;
    }
    pq[idx] = acc;
}

// ---------------------------------------------------------------------------
// Kernel 2: zero the context part of d_out (harness poisons it with 0xAA).
// ---------------------------------------------------------------------------
__global__ __launch_bounds__(256)
void zero_kernel(float* __restrict__ p, int n)
{
    const int i = blockIdx.x * 256 + threadIdx.x;
    if (i < n) p[i] = 0.0f;
}

// ---------------------------------------------------------------------------
// Kernel 3: WMMA core with TDM-staged A tiles.
//   score[b, t] = sum_u tanh(pq[b,u] + sum_m keys[b,t,m]*Wm[u,m]) * Wa[u]
//
// Grid: 2048 blocks (= B_ * T_/64), 256 threads (8 waves, wave32).
// Block tile: M = 64 t-rows, N = 512 (all of U), K = 512.
// Wave w owns N in [w*64, (w+1)*64): acc[4][4] 16x16 f32 tiles (128 VGPRs).
// A tile (keys fp32 64x32, LDS row stride 36) is DMA'd by the Tensor Data
// Mover, double-buffered; waves read it with ds_load_b128 + packed bf16 cvt
// (co-executes with the TRANS-class WMMAs). B fragments load straight from
// the L2-resident bf16 Wm.
//   A frag (ISA 16-bit A layout): lane l: m=l&15, h=l>>4;
//     element e -> K = (e<8 ? h*8+e : 16+h*8+(e-8)).
//   B frag (ISA 32x16 16-bit B layout): lane l: n=l&15, h=l>>4;
//     element e -> K = (e<8 ? h*8+e : 16+h*8+(e-8)), as for A.
// ---------------------------------------------------------------------------
__global__ __launch_bounds__(256)
void score_wmma_kernel(const float*  __restrict__ keys,
                       const __bf16* __restrict__ wmbf,
                       const float*  __restrict__ pq,
                       const float*  __restrict__ Wa,
                       float*        __restrict__ score_out)
{
    __shared__ float s_tile[2][MTILE * ROWPAD];   // 2 x 9216 B, TDM destination
    __shared__ float s_score[MTILE];

    const int b   = blockIdx.x >> 5;              // T_/64 = 32 tiles per batch
    const int t0  = (blockIdx.x & 31) * MTILE;
    const int tid = threadIdx.x;

    if (tid < MTILE) s_score[tid] = 0.0f;

    const int l      = tid & 31;
    const int w      = tid >> 5;
    const int lane16 = l & 15;
    const int h      = l >> 4;

    const float* gtile = keys + ((size_t)b * T_ + t0) * MS_;
    const unsigned lds0 = (unsigned)(uintptr_t)(void*)&s_tile[0][0];
    const unsigned lds1 = (unsigned)(uintptr_t)(void*)&s_tile[1][0];

    // Prologue: wave 0 kicks off the first tile DMA (TDM ignores EXEC; waves
    // 1..7 branch over, so exactly one tensor op is issued per block).
    if (tid < 32)
        tdm_load_keys_tile(gtile, lds0);

    v8f acc[4][4] = {};                           // [mi][ni]

    for (int kb = 0; kb < MS_; kb += KSTEP) {
        const int p = (kb >> 5) & 1;

        if (tid < 32) {
            if (kb + KSTEP < MS_) {
                tdm_load_keys_tile(gtile + kb + KSTEP, p ? lds0 : lds1);
                __builtin_amdgcn_s_wait_tensorcnt(1);   // current buffer ready
            } else {
                __builtin_amdgcn_s_wait_tensorcnt(0);
            }
        }
        __syncthreads();                           // buffer p visible to all

        // ---- A fragments from LDS (fp32 -> bf16 in registers) ----
        BF16Frag a[4];
#pragma unroll
        for (int mi = 0; mi < 4; ++mi) {
            const float* rp = &s_tile[p][(mi * 16 + lane16) * ROWPAD + h * 8];
            float4 x0 = *(const float4*)(rp);
            float4 x1 = *(const float4*)(rp + 4);
            float4 x2 = *(const float4*)(rp + 16);
            float4 x3 = *(const float4*)(rp + 20);
            a[mi].e[0]  = (__bf16)x0.x;  a[mi].e[1]  = (__bf16)x0.y;
            a[mi].e[2]  = (__bf16)x0.z;  a[mi].e[3]  = (__bf16)x0.w;
            a[mi].e[4]  = (__bf16)x1.x;  a[mi].e[5]  = (__bf16)x1.y;
            a[mi].e[6]  = (__bf16)x1.z;  a[mi].e[7]  = (__bf16)x1.w;
            a[mi].e[8]  = (__bf16)x2.x;  a[mi].e[9]  = (__bf16)x2.y;
            a[mi].e[10] = (__bf16)x2.z;  a[mi].e[11] = (__bf16)x2.w;
            a[mi].e[12] = (__bf16)x3.x;  a[mi].e[13] = (__bf16)x3.y;
            a[mi].e[14] = (__bf16)x3.z;  a[mi].e[15] = (__bf16)x3.w;
        }
        // ---- B fragments: bf16 Wm, two b128 loads per tile ----
        BF16Frag bf[4];
#pragma unroll
        for (int ni = 0; ni < 4; ++ni) {
            const __bf16* wp = wmbf + (size_t)(w * 64 + ni * 16 + lane16) * MS_ + kb + h * 8;
            bf[ni].q[0] = *(const int4*)(wp);
            bf[ni].q[1] = *(const int4*)(wp + 16);
        }
        // ---- 16 WMMAs per K step ----
#pragma unroll
        for (int mi = 0; mi < 4; ++mi)
#pragma unroll
            for (int ni = 0; ni < 4; ++ni)
                acc[mi][ni] = __builtin_amdgcn_wmma_f32_16x16x32_bf16(
                    false, a[mi].v, false, bf[ni].v,
                    (short)0, acc[mi][ni], false, false);

        __syncthreads();       // all reads of buffer p done before TDM reuses it
    }

    // ---- Epilogue: tanh(acc + pq) * Wa, reduce over u ----
    // V_TANH_F32 is a gfx1250 TRANS op: one instruction per element instead of
    // the ~30-op OCML expansion, and it co-executes with VALU.
    float pq_l[4], wa_l[4];
#pragma unroll
    for (int ni = 0; ni < 4; ++ni) {
        const int n = w * 64 + ni * 16 + lane16;
        pq_l[ni] = pq[b * U_ + n];
        wa_l[ni] = Wa[n];
    }

#pragma unroll
    for (int mi = 0; mi < 4; ++mi) {
#pragma unroll
        for (int r = 0; r < 8; ++r) {
            // C layout: lane l, VGPR r -> row m = mi*16 + h*8 + r, col = lane16
            float part = 0.0f;
#pragma unroll
            for (int ni = 0; ni < 4; ++ni)
                part += __builtin_amdgcn_tanhf(acc[mi][ni][r] + pq_l[ni]) * wa_l[ni];
            // reduce over the 16 column-lanes sharing this m
            part += __shfl_xor(part, 1);
            part += __shfl_xor(part, 2);
            part += __shfl_xor(part, 4);
            part += __shfl_xor(part, 8);
            if (lane16 == 0)
                atomicAdd(&s_score[mi * 16 + h * 8 + r], part);
        }
    }
    __syncthreads();
    if (tid < MTILE)
        score_out[(size_t)b * T_ + t0 + tid] = s_score[tid];
}

// ---------------------------------------------------------------------------
// Kernel 4: softmax over T per batch row. 64 blocks x 256 threads.
// ---------------------------------------------------------------------------
__global__ __launch_bounds__(256)
void softmax_kernel(const float* __restrict__ score, float* __restrict__ weight)
{
    __shared__ float red[256];
    const int b   = blockIdx.x;
    const int tid = threadIdx.x;
    const float* s    = score  + (size_t)b * T_;
    float*       wrow = weight + (size_t)b * T_;

    float mx = -INFINITY;
    for (int t = tid; t < T_; t += 256) mx = fmaxf(mx, s[t]);
    red[tid] = mx;
    __syncthreads();
    for (int o = 128; o > 0; o >>= 1) {
        if (tid < o) red[tid] = fmaxf(red[tid], red[tid + o]);
        __syncthreads();
    }
    mx = red[0];
    __syncthreads();

    float sum = 0.0f;
    for (int t = tid; t < T_; t += 256) {
        float e = __expf(s[t] - mx);
        wrow[t] = e;
        sum += e;
    }
    red[tid] = sum;
    __syncthreads();
    for (int o = 128; o > 0; o >>= 1) {
        if (tid < o) red[tid] += red[tid + o];
        __syncthreads();
    }
    const float inv = 1.0f / red[0];
    for (int t = tid; t < T_; t += 256) wrow[t] *= inv;
}

// ---------------------------------------------------------------------------
// Kernel 5: total_context[b,m] = sum_t weight[b,t] * keys[b,t,m]
// Split over T (16 chunks) + columns (2 chunks of 256); atomicAdd partials.
// ---------------------------------------------------------------------------
__global__ __launch_bounds__(256)
void context_kernel(const float* __restrict__ keys,
                    const float* __restrict__ weight,
                    float* __restrict__ ctx)
{
    const int b   = blockIdx.x >> 5;
    const int rem = blockIdx.x & 31;
    const int cc  = rem & 1;                   // column chunk (0..1)
    const int tc  = rem >> 1;                  // t chunk (0..15), 128 t each
    const int c   = cc * 256 + threadIdx.x;

    const float* kp = keys   + ((size_t)b * T_ + (size_t)tc * 128) * MS_ + c;
    const float* wp = weight + (size_t)b * T_ + (size_t)tc * 128;

    float acc = 0.0f;
#pragma unroll 4
    for (int t = 0; t < 128; ++t)
        acc += wp[t] * kp[(size_t)t * MS_];

    atomicAdd(&ctx[(size_t)b * MS_ + c], acc);
}

// ---------------------------------------------------------------------------
// Host launcher
// ---------------------------------------------------------------------------
extern "C" void kernel_launch(void* const* d_in, const int* in_sizes, int n_in,
                              void* d_out, int out_size, void* d_ws, size_t ws_size,
                              hipStream_t stream)
{
    const float* query = (const float*)d_in[0];   // [B, QS]
    const float* keys  = (const float*)d_in[1];   // [B, T, MS]
    const float* Wq    = (const float*)d_in[2];   // [U, QS]
    const float* Wm    = (const float*)d_in[3];   // [U, MS]
    const float* Wa    = (const float*)d_in[4];   // [1, U]

    float* ctx   = (float*)d_out;                          // [B, MS]
    float* score = (float*)d_out + (size_t)B_ * MS_;       // [B, T]

    char* ws = (char*)d_ws;
    __bf16* wmbf   = (__bf16*)ws;                              // 512 KB
    float*  pq     = (float*)(ws + 512 * 1024);                // 128 KB
    float*  weight = (float*)(ws + 512 * 1024 + 128 * 1024);   // 512 KB

    convert_wm_kernel<<<(U_ * MS_) / 256, 256, 0, stream>>>(Wm, wmbf);
    pq_kernel<<<(B_ * U_) / 256, 256, 0, stream>>>(query, Wq, pq);
    zero_kernel<<<(B_ * MS_) / 256, 256, 0, stream>>>(ctx, B_ * MS_);
    score_wmma_kernel<<<B_ * (T_ / MTILE), 256, 0, stream>>>(keys, wmbf, pq, Wa, score);
    softmax_kernel<<<B_, 256, 0, stream>>>(score, weight);
    context_kernel<<<B_ * 32, 256, 0, stream>>>(keys, weight, ctx);
}
